// WaveglowBlock_26414048871078
// MI455X (gfx1250) — matrix operand order repaired and checked by reading, not compiled, not run on hardware
//
#include <hip/hip_runtime.h>

typedef __attribute__((ext_vector_type(8)))  float    v8f;
typedef __attribute__((ext_vector_type(8)))  _Float16 v8h;
typedef __attribute__((ext_vector_type(16))) _Float16 v16h;

#define ROWS    65536          // B*T = 16*4096
#define TT      4096
#define NCH     256
#define NMEL    640
#define KMAIN   1408           // 3*256 (conv taps) + 640 (cond)
#define KC_MAIN 44             // 1408/32
#define NT_MAIN 32             // 512/16
#define MROWS   64             // rows per block (4 M-tiles per wave)

static __device__ __forceinline__ v8f wmma_f16(v16h a, v16h b, v8f c) {
  return __builtin_amdgcn_wmma_f32_16x16x32_f16(false, a, false, b, (short)0, c,
                                                false, false);
}

// ---------------------------------------------------------------------------
// Repack an fp32 [K][N] weight (optionally concatenated from two sources along
// K: src0 rows [0,K0), src1 rows [K0,K)) into f16 WMMA B fragments.
// Fragment layout per ISA 7.12.2 (B 32x16 f16, wave32):
//   lane = khalf*16 + n, halfs h=0..7 -> K=8*khalf+h, h=8..15 -> K=16+8*khalf+(h-8)
//   packed address (halfs): ((c*NT + nt)*32 + lane)*16 + h   -> 32B/lane
// ---------------------------------------------------------------------------
__global__ void pack_b_kernel(const float* __restrict__ src0, int K0,
                              const float* __restrict__ src1, int N,
                              int totalDwords, unsigned* __restrict__ dst) {
  int q = blockIdx.x * 256 + threadIdx.x;
  if (q >= totalDwords) return;
  int d    = q & 7;
  int lane = (q >> 3) & 31;
  int f    = q >> 8;
  int NT   = N >> 4;
  int nt   = f % NT;
  int c    = f / NT;
  int n    = nt * 16 + (lane & 15);
  int kb   = c * 32 + ((d >> 2) << 4) + ((lane >> 4) << 3) + (d & 3) * 2;   // even; same side of K0
  const float* r0 = (kb < K0) ? (src0 + (size_t)kb * N)
                              : (src1 + (size_t)(kb - K0) * N);
  const float* r1 = (kb < K0) ? (src0 + (size_t)(kb + 1) * N)
                              : (src1 + (size_t)(kb + 1 - K0) * N);
  union { _Float16 h[2]; unsigned u; } p;
  p.h[0] = (_Float16)r0[n];
  p.h[1] = (_Float16)r1[n];
  dst[q] = p.u;
}

// spect f32 -> f16 (once; saves 0.65 GB of staging traffic across 8 layers)
__global__ void cvt_spect_kernel(const float* __restrict__ src,
                                 _Float16* __restrict__ dst) {
  size_t base = ((size_t)blockIdx.x * 256 + threadIdx.x) * 8;
  const float4* p = (const float4*)(src + base);
  float4 q0 = p[0], q1 = p[1];
  union { _Float16 h[8]; uint4 u; } pk;
  pk.h[0]=(_Float16)q0.x; pk.h[1]=(_Float16)q0.y; pk.h[2]=(_Float16)q0.z;
  pk.h[3]=(_Float16)q0.w; pk.h[4]=(_Float16)q1.x; pk.h[5]=(_Float16)q1.y;
  pk.h[6]=(_Float16)q1.z; pk.h[7]=(_Float16)q1.w;
  *(uint4*)(dst + base) = pk.u;
}

// h = audio @ start_w + start_b   (K=4: scalar, memory bound) -> f16
__global__ void start_kernel(const float* __restrict__ audio,
                             const float* __restrict__ w,
                             const float* __restrict__ b,
                             _Float16* __restrict__ h) {
  int idx = blockIdx.x * 256 + threadIdx.x;
  int row = idx >> 8;
  int cc  = idx & 255;
  const float* a = audio + (size_t)row * 4;
  float s = b[cc];
#pragma unroll
  for (int j = 0; j < 4; ++j) s += a[j] * w[j * 256 + cc];
  h[idx] = (_Float16)s;
}

// out = out_acc @ end_w + end_b   (N=8: scalar, memory bound)
__global__ void end_kernel(const float* __restrict__ oacc,
                           const float* __restrict__ w,
                           const float* __restrict__ b,
                           float* __restrict__ out) {
  int idx = blockIdx.x * 256 + threadIdx.x;
  int row = idx >> 3;
  int o   = idx & 7;
  const float* r = oacc + (size_t)row * 256;
  float s = b[o];
#pragma unroll 8
  for (int k = 0; k < 256; ++k) s += r[k] * w[k * 8 + o];
  out[idx] = s;
}

// ---------------------------------------------------------------------------
// Fused WaveNet layer. One block = 64 time-rows x full 512 cols.
// Wave tile = 64 rows x 64 cols (4M x 4N WMMA tiles) => B fragments reused
// across 4 M-tiles in registers: 64 FLOP per L2 byte of weight traffic.
// Phase 1: x = dilconv(h) + spect@cond_w  (K=1408 WMMA, f16 A double-buffered
//          through LDS: one barrier per chunk, stage load hidden under WMMA)
// Epilogue: gate in 16-row slices -> acts (f16, LDS resident for all 64 rows)
// Phase 2: rs = acts @ rs_w (K=256 WMMA, 4Mx4N) -> h/skip update.
// ---------------------------------------------------------------------------
__global__ __launch_bounds__(256) void layer_kernel(
    const _Float16* __restrict__ h_in, _Float16* __restrict__ h_out,
    const _Float16* __restrict__ spect16,
    const _Float16* __restrict__ pB,     // packed 1408x512
    const _Float16* __restrict__ pBrs,   // packed 256x512 (or 256x256 last)
    const float* __restrict__ in_b, const float* __restrict__ cond_b,
    const float* __restrict__ rs_b,
    float* __restrict__ out_acc,
    int dil, int first, int last) {
  __shared__ __align__(16) _Float16 ldsA[2][MROWS][40];  // 64x32 A (+pad) x2
  __shared__ __align__(16) _Float16 ldsX[16][520];       // 16x512 x slice
  __shared__ __align__(16) _Float16 ldsAct[MROWS][264];  // 64x256 acts

  const int tid  = threadIdx.x;
  const int wv   = tid >> 5;
  const int lane = tid & 31;
  const int r0   = blockIdx.x << 6;
  const int bb   = r0 >> 12;         // batch index (64 | 4096: no crossing)
  const int t0   = r0 & 4095;

  // A-fragment addressing (16x32 f16): row = lane&15,
  // halfs 0..7 -> K=koff..koff+7, halfs 8..15 -> K=koff+16..koff+23.
  const int koff = (lane >> 4) << 3;
  const int arow = lane & 15;
  // staging: thread -> 8 consecutive halfs (16B) of the 64x32 chunk
  const int sr = tid >> 2;
  const int sk = (tid & 3) << 3;

  const v16h* Bf = (const v16h*)pB;

  // stage-load for K-chunk c: pure 16B f16 copy, zero-filled at conv borders
  auto stage_load = [&](int c) -> uint4 {
    if (c < 24) {                      // conv taps: K index = tap*256 + ch
      int tap = c >> 3;
      int kk  = (c & 7) << 5;
      int t   = t0 + sr + (tap - 1) * dil;
      if (t < 0 || t >= TT) { uint4 z = {0u, 0u, 0u, 0u}; return z; }
      return *(const uint4*)(h_in + (((size_t)(bb * TT + t)) << 8) + kk + sk);
    }
    int kk = (c - 24) << 5;            // conditioning: spect channels
    return *(const uint4*)(spect16 + (size_t)(bb * TT + t0 + sr) * NMEL + kk + sk);
  };

  v8f acc[4][4] = {};   // [m][n] 16 tiles, 128 VGPRs

  {
    uint4 s0 = stage_load(0);
    *(uint4*)&ldsA[0][sr][sk] = s0;
  }
  __syncthreads();

  for (int c = 0; c < KC_MAIN; ++c) {
    const int cur = c & 1;
    uint4 nxt;
    const bool have = (c + 1 < KC_MAIN);
    if (have) nxt = stage_load(c + 1);   // overlaps with WMMAs below

    v16h a[4];
#pragma unroll
    for (int m = 0; m < 4; ++m) {
      const v8h lo = *(const v8h*)&ldsA[cur][m * 16 + arow][koff];
      const v8h hi = *(const v8h*)&ldsA[cur][m * 16 + arow][koff + 16];
#pragma unroll
      for (int j = 0; j < 8; ++j) { a[m][j] = lo[j]; a[m][8 + j] = hi[j]; }
    }
    size_t fbase = ((size_t)c * NT_MAIN + (wv << 2)) * 32 + lane;
    if (have)
      __builtin_prefetch(&Bf[fbase + (size_t)NT_MAIN * 32], 0, 1);
#pragma unroll
    for (int n = 0; n < 4; ++n) {
      v16h b = Bf[fbase + (size_t)n * 32];
#pragma unroll
      for (int m = 0; m < 4; ++m) acc[m][n] = wmma_f16(a[m], b, acc[m][n]);
    }
    if (have) *(uint4*)&ldsA[cur ^ 1][sr][sk] = nxt;
    __syncthreads();
  }

  // Epilogue: per 16-row slice, gather x across waves and gate.
  // C layout: lanes 0-15 -> M=j, N=lane; lanes 16-31 -> M=j+8, N=lane-16
  const int rb = (lane >> 4) << 3;
  const int cb = (wv << 6) + (lane & 15);
#pragma unroll
  for (int m = 0; m < 4; ++m) {
    __syncthreads();
#pragma unroll
    for (int n = 0; n < 4; ++n)
#pragma unroll
      for (int j = 0; j < 8; ++j)
        ldsX[rb + j][cb + n * 16] = (_Float16)acc[m][n][j];
    __syncthreads();
    {
      int row = tid >> 4;
      int c0  = (tid & 15) << 4;
#pragma unroll
      for (int u = 0; u < 16; ++u) {
        int cc = c0 + u;
        float xt = (float)ldsX[row][cc]       + in_b[cc]       + cond_b[cc];
        float xs = (float)ldsX[row][cc + 256] + in_b[cc + 256] + cond_b[cc + 256];
        float g  = tanhf(xt) * (1.0f / (1.0f + __expf(-xs)));
        ldsAct[m * 16 + row][cc] = (_Float16)g;
      }
    }
  }
  __syncthreads();

  // Phase 2: rs = acts @ rs_w (K=256). Last layer: N=256 -> waves 0..3 only.
  const int NT2 = last ? 16 : 32;
  if ((wv << 2) < NT2) {
    v8f s[4][4] = {};
    const v16h* B2 = (const v16h*)pBrs;
    for (int c2 = 0; c2 < 8; ++c2) {
      v16h a[4];
#pragma unroll
      for (int m = 0; m < 4; ++m) {
        const v8h lo = *(const v8h*)&ldsAct[m * 16 + arow][(c2 << 5) + koff];
        const v8h hi = *(const v8h*)&ldsAct[m * 16 + arow][(c2 << 5) + koff + 16];
#pragma unroll
        for (int j = 0; j < 8; ++j) { a[m][j] = lo[j]; a[m][8 + j] = hi[j]; }
      }
      size_t fb = ((size_t)c2 * NT2 + (wv << 2)) * 32 + lane;
#pragma unroll
      for (int n = 0; n < 4; ++n) {
        v16h b = B2[fb + (size_t)n * 32];
#pragma unroll
        for (int m = 0; m < 4; ++m) s[m][n] = wmma_f16(a[m], b, s[m][n]);
      }
    }

#pragma unroll
    for (int m = 0; m < 4; ++m) {
#pragma unroll
      for (int n = 0; n < 4; ++n) {
        int col = (wv << 6) + (n << 4) + (lane & 15);
        float bias = rs_b[col];
        int rowbase = r0 + m * 16 + rb;
        if (!last) {
          if (col < NCH) {        // residual: h_out = rs[:, :256] + h_in
#pragma unroll
            for (int j = 0; j < 8; ++j) {
              size_t idx = ((size_t)(rowbase + j) << 8) + col;
              h_out[idx] = (_Float16)(s[m][n][j] + bias + (float)h_in[idx]);
            }
          } else {                // skip accumulation (f32)
            int oc = col - NCH;
#pragma unroll
            for (int j = 0; j < 8; ++j) {
              size_t idx = ((size_t)(rowbase + j) << 8) + oc;
              float prev = first ? 0.0f : out_acc[idx];
              out_acc[idx] = prev + s[m][n][j] + bias;
            }
          }
        } else {                  // last layer: skip only (col in 0..255)
#pragma unroll
          for (int j = 0; j < 8; ++j) {
            size_t idx = ((size_t)(rowbase + j) << 8) + col;
            out_acc[idx] = out_acc[idx] + s[m][n][j] + bias;
          }
        }
      }
    }
  }
}

extern "C" void kernel_launch(void* const* d_in, const int* in_sizes, int n_in,
                              void* d_out, int out_size, void* d_ws, size_t ws_size,
                              hipStream_t stream) {
  (void)in_sizes; (void)n_in; (void)out_size; (void)ws_size;
  const float* audio     = (const float*)d_in[0];
  const float* spect     = (const float*)d_in[1];
  const float* start_w   = (const float*)d_in[2];
  const float* start_b   = (const float*)d_in[3];
  const float* in_w      = (const float*)d_in[4];   // (8,3,256,512)
  const float* in_b      = (const float*)d_in[5];   // (8,512)
  const float* cond_w    = (const float*)d_in[6];   // (8,640,512)
  const float* cond_b    = (const float*)d_in[7];   // (8,512)
  const float* rs_w      = (const float*)d_in[8];   // (7,256,512)
  const float* rs_b      = (const float*)d_in[9];   // (7,512)
  const float* rs_w_last = (const float*)d_in[10];  // (256,256)
  const float* rs_b_last = (const float*)d_in[11];  // (256,)
  const float* end_w     = (const float*)d_in[12];  // (256,8)
  const float* end_b     = (const float*)d_in[13];  // (8,)
  float* out = (float*)d_out;

  // workspace layout
  _Float16* hA      = (_Float16*)d_ws;                     // 32 MB
  _Float16* hB      = hA + (size_t)ROWS * NCH;             // 32 MB
  float*    oacc    = (float*)(hB + (size_t)ROWS * NCH);   // 64 MB
  _Float16* spect16 = (_Float16*)(oacc + (size_t)ROWS * NCH);   // 80 MB
  _Float16* pmain   = spect16 + (size_t)ROWS * NMEL;       // 11.5 MB
  _Float16* prs     = pmain + (size_t)8 * KMAIN * 512;     // 1.75 MB
  _Float16* prsl    = prs + (size_t)7 * 256 * 512;         // 128 KB

  // 1) pack weights into WMMA B-fragment layout (f16); convert spect to f16
  for (int i = 0; i < 8; ++i) {
    int totalDw = KMAIN * 512 / 2;                     // 360448
    pack_b_kernel<<<(totalDw + 255) / 256, 256, 0, stream>>>(
        in_w + (size_t)i * 3 * 256 * 512, 768,
        cond_w + (size_t)i * 640 * 512, 512, totalDw,
        (unsigned*)(pmain + (size_t)i * KMAIN * 512));
  }
  for (int i = 0; i < 7; ++i) {
    int totalDw = 256 * 512 / 2;                       // 65536
    pack_b_kernel<<<(totalDw + 255) / 256, 256, 0, stream>>>(
        rs_w + (size_t)i * 256 * 512, 256, rs_w, 512, totalDw,
        (unsigned*)(prs + (size_t)i * 256 * 512));
  }
  {
    int totalDw = 256 * 256 / 2;                       // 32768
    pack_b_kernel<<<(totalDw + 255) / 256, 256, 0, stream>>>(
        rs_w_last, 256, rs_w_last, 256, totalDw, (unsigned*)prsl);
  }
  cvt_spect_kernel<<<((size_t)ROWS * NMEL / 8) / 256, 256, 0, stream>>>(
      spect, spect16);

  // 2) start projection
  start_kernel<<<(ROWS * NCH) / 256, 256, 0, stream>>>(audio, start_w, start_b, hA);

  // 3) fused layers
  for (int i = 0; i < 8; ++i) {
    const _Float16* hin  = (i & 1) ? hB : hA;
    _Float16*       hout = (i & 1) ? hA : hB;
    int last  = (i == 7);
    int first = (i == 0);
    const _Float16* pB   = pmain + (size_t)i * KMAIN * 512;
    const _Float16* pBrs = last ? prsl : (prs + (size_t)i * 256 * 512);
    const float*    rsb  = last ? rs_b_last : (rs_b + (size_t)i * 512);
    layer_kernel<<<ROWS / MROWS, 256, 0, stream>>>(
        hin, hout, spect16, pB, pBrs,
        in_b + (size_t)i * 512, cond_b + (size_t)i * 512, rsb,
        oacc, 1 << i, first, last);
  }

  // 4) end projection
  end_kernel<<<(ROWS * 8) / 256, 256, 0, stream>>>(oacc, end_w, end_b, out);
}
